// GNN_v2_28303834480935
// MI455X (gfx1250) — matrix operand order repaired and checked by reading, not compiled, not run on hardware
//
#include <hip/hip_runtime.h>

typedef __attribute__((ext_vector_type(16))) _Float16 v16h;
typedef __attribute__((ext_vector_type(8)))  _Float16 v8h;
typedef __attribute__((ext_vector_type(8)))  float    v8f;

#define NB   16
#define NN   256
#define DD   32
#define PTOT (16.0f * 256.0f * 256.0f)
#define LEAK 0.01f
#define BNEPS 1e-5f

// workspace layout (float offsets)
#define WS_H    0                   // 16*256*32            = 131072
#define WS_PART 131072              // 4096 * 128           = 524288
#define WS_FOLD (131072 + 524288)   // 4 layers * 128       = 512

struct SH {
  _Float16 hb[NN][DD];                 // h[b][j][d] in f16 (16KB)
  float    hi32[DD];                   // h[b][i][:] in f32
  alignas(32) _Float16 w1[64][32];     // W^T[cout][cin], layer 1 (32->64)
  alignas(32) _Float16 w2[64][64];     // layer 2 (64->64)
  alignas(32) _Float16 w3[32][64];     // layer 3 (64->32)
  alignas(32) _Float16 w4[32][32];     // layer 4 (32->32)
  float fa[4][64];                     // folded BN scale  per layer/channel
  float fb[4][64];                     // folded BN shift
  float bconv[64];                     // conv bias of current stats layer
  float outb_s;
  alignas(32) _Float16 stage[8][16][72]; // per-wave transpose staging (pad 72 vs banks)
  union {
    struct { float red[8][64]; float red2[8][64]; } st;          // stats reduction
    struct { float logits[NN]; float probs[NN]; float sred[NN];
             float aggred[8][DD]; float aggv[DD];
             float wout[DD]; float lin[DD][DD]; float linb[DD]; } fi; // final pass
  } u;
};

// scalar |x| via explicit VALU op; opaque to SLP/InstCombine (works around
// gfx1250 ISel crash on vectorized fabs-of-packed-fadd patterns)
__device__ __forceinline__ float abs_asm(float x) {
  float r;
  asm("v_and_b32 %0, 0x7fffffff, %1" : "=v"(r) : "v"(x));
  return r;
}

// ---------------- h builder: x[N,bsz,dm] -> h[b][n][d] --------------------
__global__ void __launch_bounds__(256) k_build_h(const float* __restrict__ x,
                                                 float* __restrict__ h) {
  int idx = blockIdx.x * 256 + threadIdx.x;   // 131072 total
  int d = idx & 31;
  int p = (idx >> 5) & 255;                   // n
  int q = idx >> 13;                          // b
  int s = p >> 7;
  int np = 2 * (p & 127) + (q >> 3);
  int c = (q & 7) * 32 + d;
  h[(q * NN + p) * DD + d] = x[np * 512 + s * 256 + c];
}

// -------- one folded conv layer: Ain (CIN) -> Aout (COUT), via WMMA -------
template <int CIN, int COUT>
__device__ __forceinline__ void conv_fold_step(SH* sh, const _Float16* wt, int l,
                                               const v16h* Ain, v16h* Aout,
                                               _Float16 (*stg)[72], int lane) {
  const int hf = lane >> 4, nl = lane & 15;
  __builtin_amdgcn_wave_barrier();
#pragma unroll
  for (int nt = 0; nt < COUT / 16; ++nt) {
    v8f acc = {0.f, 0.f, 0.f, 0.f, 0.f, 0.f, 0.f, 0.f};
#pragma unroll
    for (int kt = 0; kt < CIN / 32; ++kt) {
      const _Float16* bp = wt + (nt * 16 + nl) * CIN + kt * 32 + hf * 8;
      const v16h B = __builtin_shufflevector(*(const v8h*)bp, *(const v8h*)(bp + 16),
                                             0, 1, 2, 3, 4, 5, 6, 7, 8, 9, 10, 11, 12, 13, 14, 15);
      acc = __builtin_amdgcn_wmma_f32_16x16x32_f16(false, Ain[kt], false, B,
                                                   (short)0, acc, false, false);
    }
    const int c = nt * 16 + nl;
    const float al = sh->fa[l][c], be = sh->fb[l][c];
#pragma unroll
    for (int r = 0; r < 8; ++r) {
      float z = fmaf(acc[r], al, be);
      z = z > 0.f ? z : LEAK * z;                    // LeakyReLU
      stg[r + hf * 8][c] = (_Float16)z;              // C-layout -> [pos][chan]
    }
  }
  __builtin_amdgcn_wave_barrier();
  asm volatile("s_wait_dscnt 0" ::: "memory");       // same-wave LDS RAW guard
  __builtin_amdgcn_wave_barrier();
#pragma unroll
  for (int kt = 0; kt < COUT / 32; ++kt) {           // repack into A-layout
    v8h lo = *(const v8h*)&stg[nl][kt * 32 + hf * 8];
    v8h hi = *(const v8h*)&stg[nl][kt * 32 + 16 + hf * 8];
    Aout[kt] = __builtin_shufflevector(lo, hi, 0, 1, 2, 3, 4, 5, 6, 7,
                                       8, 9, 10, 11, 12, 13, 14, 15);
  }
}

// ----- terminal conv of a stats pass: accumulate sum / sumsq of y --------
template <int CIN, int COUT>
__device__ __forceinline__ void conv_stats_step(SH* sh, const _Float16* wt,
                                                const v16h* Ain, float* ssum,
                                                float* ssq, int lane) {
  const int hf = lane >> 4, nl = lane & 15;
#pragma unroll
  for (int nt = 0; nt < COUT / 16; ++nt) {
    v8f acc = {0.f, 0.f, 0.f, 0.f, 0.f, 0.f, 0.f, 0.f};
#pragma unroll
    for (int kt = 0; kt < CIN / 32; ++kt) {
      const _Float16* bp = wt + (nt * 16 + nl) * CIN + kt * 32 + hf * 8;
      const v16h B = __builtin_shufflevector(*(const v8h*)bp, *(const v8h*)(bp + 16),
                                             0, 1, 2, 3, 4, 5, 6, 7, 8, 9, 10, 11, 12, 13, 14, 15);
      acc = __builtin_amdgcn_wmma_f32_16x16x32_f16(false, Ain[kt], false, B,
                                                   (short)0, acc, false, false);
    }
    const float bc = sh->bconv[nt * 16 + nl];
#pragma unroll
    for (int r = 0; r < 8; ++r) {
      float v = acc[r] + bc;
      ssum[nt] += v;
      ssq[nt] += v * v;
    }
  }
}

// ------------- chain for one 16-position tile of row (b,i) ---------------
template <int DEPTH>
__device__ __forceinline__ void tile_chain(SH* sh, int i, int j0, int lane,
                                           _Float16 (*stg)[72], float* ssum,
                                           float* ssq, float outb) {
  const int hf = lane >> 4, nl = lane & 15;
  v16h A[2];
  {  // build a0 = |h_i - h_j| directly in WMMA A-fragment layout
    const int j = j0 + nl;
#pragma unroll
    for (int e = 0; e < 16; ++e) {
      const int k = ((e >> 3) << 4) + (e & 7) + (hf << 3);
      const float d = sh->hi32[k] - (float)sh->hb[j][k];
      A[0][e] = (_Float16)abs_asm(d);
    }
  }
  if (DEPTH == 1) { conv_stats_step<32, 64>(sh, &sh->w1[0][0], A, ssum, ssq, lane); return; }
  conv_fold_step<32, 64>(sh, &sh->w1[0][0], 0, A, A, stg, lane);
  if (DEPTH == 2) { conv_stats_step<64, 64>(sh, &sh->w2[0][0], A, ssum, ssq, lane); return; }
  conv_fold_step<64, 64>(sh, &sh->w2[0][0], 1, A, A, stg, lane);
  if (DEPTH == 3) { conv_stats_step<64, 32>(sh, &sh->w3[0][0], A, ssum, ssq, lane); return; }
  conv_fold_step<64, 32>(sh, &sh->w3[0][0], 2, A, A, stg, lane);
  if (DEPTH == 4) { conv_stats_step<32, 32>(sh, &sh->w4[0][0], A, ssum, ssq, lane); return; }
  conv_fold_step<32, 32>(sh, &sh->w4[0][0], 3, A, A, stg, lane);
  if (lane < 16) {  // logits from z4 in staging tile
    float lg = outb;
#pragma unroll
    for (int c = 0; c < 32; ++c) lg += (float)stg[nl][c] * sh->u.fi.wout[c];
    sh->u.fi.logits[j0 + nl] = lg;
  }
}

// ------------------------- main fused kernel -----------------------------
template <int DEPTH>  // 1..4: stats pass for layer DEPTH; 5: final pass
__global__ void __launch_bounds__(256) k_chain(
    const float* __restrict__ h, const float* __restrict__ w1g,
    const float* __restrict__ w2g, const float* __restrict__ w3g,
    const float* __restrict__ w4g, const float* __restrict__ bcur,
    const float* __restrict__ fold, const float* __restrict__ outw,
    const float* __restrict__ outbp, const float* __restrict__ linw,
    const float* __restrict__ linbp, float* __restrict__ part,
    float* __restrict__ outbuf, float* __restrict__ Abuf) {
  __shared__ SH sh;
  const int tid = threadIdx.x;
  const int bb = blockIdx.x >> 8;
  const int i = blockIdx.x & 255;

  for (int idx = tid; idx < NN * DD; idx += 256)
    sh.hb[idx >> 5][idx & 31] = (_Float16)h[(size_t)bb * NN * DD + idx];
  if (tid < DD) sh.hi32[tid] = h[((size_t)bb * NN + i) * DD + tid];

  // weights transposed to [cout][cin], f16
  for (int idx = tid; idx < 64 * 32; idx += 256) { int n = idx >> 5, k = idx & 31; sh.w1[n][k] = (_Float16)w1g[k * 64 + n]; }
  for (int idx = tid; idx < 64 * 64; idx += 256) { int n = idx >> 6, k = idx & 63; sh.w2[n][k] = (_Float16)w2g[k * 64 + n]; }
  for (int idx = tid; idx < 32 * 64; idx += 256) { int n = idx >> 6, k = idx & 63; sh.w3[n][k] = (_Float16)w3g[k * 32 + n]; }
  for (int idx = tid; idx < 32 * 32; idx += 256) { int n = idx >> 5, k = idx & 31; sh.w4[n][k] = (_Float16)w4g[k * 32 + n]; }

  const int nfold = ((DEPTH == 5) ? 4 : (DEPTH - 1)) * 128;
  for (int idx = tid; idx < nfold; idx += 256) {
    int l = idx >> 7, r = idx & 127;
    if (r < 64) sh.fa[l][r] = fold[idx]; else sh.fb[l][r - 64] = fold[idx];
  }
  if (DEPTH <= 4) { const int cl = (DEPTH <= 2) ? 64 : 32; if (tid < cl) sh.bconv[tid] = bcur[tid]; }
  if (DEPTH == 5) {
    if (tid < DD) { sh.u.fi.wout[tid] = outw[tid]; sh.u.fi.linb[tid] = linbp[tid]; }
    for (int idx = tid; idx < DD * DD; idx += 256) sh.u.fi.lin[idx >> 5][idx & 31] = linw[idx];
    if (tid == 0) sh.outb_s = outbp[0];
  }
  __syncthreads();

  const int wave = tid >> 5, lane = tid & 31;
  _Float16(*stg)[72] = sh.stage[wave];
  float ssum[4] = {0.f, 0.f, 0.f, 0.f}, ssq[4] = {0.f, 0.f, 0.f, 0.f};
  const float outb = (DEPTH == 5) ? sh.outb_s : 0.f;

#pragma unroll
  for (int t = 0; t < 2; ++t)  // wave handles 32 j's = two 16-pos tiles
    tile_chain<DEPTH>(&sh, i, wave * 32 + t * 16, lane, stg, ssum, ssq, outb);

  if (DEPTH <= 4) {  // deterministic per-workgroup stats partials
    const int NT = (DEPTH <= 2) ? 4 : 2;
#pragma unroll
    for (int nt = 0; nt < NT; ++nt) {
      ssum[nt] += __shfl_xor(ssum[nt], 16);
      ssq[nt] += __shfl_xor(ssq[nt], 16);
    }
    if (lane < 16)
      for (int nt = 0; nt < NT; ++nt) {
        sh.u.st.red[wave][nt * 16 + lane] = ssum[nt];
        sh.u.st.red2[wave][nt * 16 + lane] = ssq[nt];
      }
    __syncthreads();
    if (tid < NT * 16) {
      float S = 0.f, Q = 0.f;
#pragma unroll
      for (int w = 0; w < 8; ++w) { S += sh.u.st.red[w][tid]; Q += sh.u.st.red2[w][tid]; }
      part[(size_t)blockIdx.x * 128 + tid] = S;
      part[(size_t)blockIdx.x * 128 + 64 + tid] = Q;
    }
  } else {  // final: softmax over the 256 j's of this row, then agg + linear
    __syncthreads();
    const float lg = sh.u.fi.logits[tid];
    sh.u.fi.sred[tid] = lg;
    __syncthreads();
    for (int off = 128; off > 0; off >>= 1) {
      if (tid < off) sh.u.fi.sred[tid] = fmaxf(sh.u.fi.sred[tid], sh.u.fi.sred[tid + off]);
      __syncthreads();
    }
    const float mx = sh.u.fi.sred[0];
    __syncthreads();
    const float ev = __expf(lg - mx);
    sh.u.fi.sred[tid] = ev;
    __syncthreads();
    for (int off = 128; off > 0; off >>= 1) {
      if (tid < off) sh.u.fi.sred[tid] += sh.u.fi.sred[tid + off];
      __syncthreads();
    }
    const float pv = ev / sh.u.fi.sred[0];
    sh.u.fi.probs[tid] = pv;
    Abuf[((size_t)bb * NN + i) * NN + tid] = pv;
    __syncthreads();
    {  // agg[d] = sum_j p[j]*h[b][j][d], split j into 8 chunks (deterministic)
      const int d = tid & 31, ch = tid >> 5;
      float p = 0.f;
      for (int jj = 0; jj < 32; ++jj)
        p += sh.u.fi.probs[ch * 32 + jj] * (float)sh.hb[ch * 32 + jj][d];
      sh.u.fi.aggred[ch][d] = p;
    }
    __syncthreads();
    if (tid < DD) {
      float a = 0.f;
      for (int ch = 0; ch < 8; ++ch) a += sh.u.fi.aggred[ch][tid];
      sh.u.fi.aggv[tid] = a;
    }
    __syncthreads();
    if (tid < DD) {
      float o = sh.u.fi.linb[tid];
      for (int d = 0; d < 32; ++d) o += sh.u.fi.aggv[d] * sh.u.fi.lin[tid][d];
      outbuf[((size_t)bb * NN + i) * DD + tid] = o;
    }
  }
}

// ---- reduce 4096 partials per channel, fold BN+bias into affine ---------
__global__ void __launch_bounds__(256) k_reduce_fold(
    const float* __restrict__ part, const float* __restrict__ gamma,
    const float* __restrict__ bnbeta, const float* __restrict__ bconv,
    float* __restrict__ fold, int l) {
  __shared__ float rs[256], rq[256];
  const int c = blockIdx.x, tid = threadIdx.x;
  float S = 0.f, Q = 0.f;
  for (int w = tid; w < 4096; w += 256) {
    S += part[(size_t)w * 128 + c];
    Q += part[(size_t)w * 128 + 64 + c];
  }
  rs[tid] = S; rq[tid] = Q;
  __syncthreads();
  for (int off = 128; off > 0; off >>= 1) {
    if (tid < off) { rs[tid] += rs[tid + off]; rq[tid] += rq[tid + off]; }
    __syncthreads();
  }
  if (tid == 0) {
    const float mean = rs[0] * (1.f / PTOT);
    const float var = rq[0] * (1.f / PTOT) - mean * mean;
    const float al = gamma[c] * rsqrtf(var + BNEPS);
    fold[l * 128 + c] = al;
    fold[l * 128 + 64 + c] = (bconv[c] - mean) * al + bnbeta[c];
  }
}

extern "C" void kernel_launch(void* const* d_in, const int* in_sizes, int n_in,
                              void* d_out, int out_size, void* d_ws, size_t ws_size,
                              hipStream_t stream) {
  const float* x   = (const float*)d_in[0];
  const float* w1  = (const float*)d_in[1];
  const float* w2  = (const float*)d_in[2];
  const float* w3  = (const float*)d_in[3];
  const float* w4  = (const float*)d_in[4];
  const float* b1  = (const float*)d_in[5];
  const float* b2  = (const float*)d_in[6];
  const float* b3  = (const float*)d_in[7];
  const float* b4  = (const float*)d_in[8];
  const float* g1  = (const float*)d_in[9];
  const float* g2  = (const float*)d_in[10];
  const float* g3  = (const float*)d_in[11];
  const float* g4  = (const float*)d_in[12];
  const float* be1 = (const float*)d_in[13];
  const float* be2 = (const float*)d_in[14];
  const float* be3 = (const float*)d_in[15];
  const float* be4 = (const float*)d_in[16];
  const float* outw = (const float*)d_in[17];
  const float* outb = (const float*)d_in[18];
  const float* linw = (const float*)d_in[19];
  const float* linb = (const float*)d_in[20];

  float* ws = (float*)d_ws;
  float* h = ws + WS_H;
  float* part = ws + WS_PART;
  float* fold = ws + WS_FOLD;
  float* outbuf = (float*)d_out;                 // [16,256,32]
  float* Abuf = outbuf + (size_t)NB * NN * DD;   // [16,256,256]

  k_build_h<<<dim3(512), dim3(256), 0, stream>>>(x, h);

  k_chain<1><<<dim3(4096), dim3(256), 0, stream>>>(h, w1, w2, w3, w4, b1, fold,
                                                   outw, outb, linw, linb, part, outbuf, Abuf);
  k_reduce_fold<<<dim3(64), dim3(256), 0, stream>>>(part, g1, be1, b1, fold, 0);

  k_chain<2><<<dim3(4096), dim3(256), 0, stream>>>(h, w1, w2, w3, w4, b2, fold,
                                                   outw, outb, linw, linb, part, outbuf, Abuf);
  k_reduce_fold<<<dim3(64), dim3(256), 0, stream>>>(part, g2, be2, b2, fold, 1);

  k_chain<3><<<dim3(4096), dim3(256), 0, stream>>>(h, w1, w2, w3, w4, b3, fold,
                                                   outw, outb, linw, linb, part, outbuf, Abuf);
  k_reduce_fold<<<dim3(32), dim3(256), 0, stream>>>(part, g3, be3, b3, fold, 2);

  k_chain<4><<<dim3(4096), dim3(256), 0, stream>>>(h, w1, w2, w3, w4, b4, fold,
                                                   outw, outb, linw, linb, part, outbuf, Abuf);
  k_reduce_fold<<<dim3(32), dim3(256), 0, stream>>>(part, g4, be4, b4, fold, 3);

  k_chain<5><<<dim3(4096), dim3(256), 0, stream>>>(h, w1, w2, w3, w4, b1, fold,
                                                   outw, outb, linw, linb, part, outbuf, Abuf);
}
